// Mamba2SingleScan_6167573037312
// MI455X (gfx1250) — matrix operand order repaired and checked by reading, not compiled, not run on hardware
//
#include <hip/hip_runtime.h>
#include <hip/hip_bf16.h>

// ---------------- problem constants (from reference) ----------------
#define BD      2
#define LD      4096
#define DINNER  2048
#define DSTATE  128
#define DCONV   4
#define HEADDIM 64
#define NHEADS  32
#define CHUNK   256
#define NCHUNK  (LD / CHUNK)                 // 16
#define CONVDIM (DINNER + 2 * DSTATE)        // 2304
#define DIN     (2 * DINNER + 2 * DSTATE + NHEADS) // 4384
#define EPSV    1e-5f

typedef __bf16 bf16_t;
typedef __attribute__((ext_vector_type(16))) __bf16 v16bf;
typedef __attribute__((ext_vector_type(8)))  __bf16 v8bf;
typedef __attribute__((ext_vector_type(8)))  float  v8f;
typedef int v4i_ __attribute__((vector_size(16)));   // matches builtin's int4 pointee

// LDS leading dims: elems ≡ 8 (mod 128) -> byte stride ≡ 16 (mod 256) -> lane
// b128 accesses land on disjoint 4-bank groups (conflict-free).
#define NLD  (DSTATE + 8)    // 136 bf16 (272 B)
#define KLD  (CHUNK + 8)     // 264 bf16 (528 B)
#define GLD  40              // 32 + 8 bf16 (80 B)

#define AS1 __attribute__((address_space(1)))
#define AS3 __attribute__((address_space(3)))

// gfx1250 async global->LDS copies (ASYNCcnt-tracked); guarded so either
// toolchain arity/availability still compiles (falls back to b128 copy).
#if defined(__AMDGCN__) && __has_builtin(__builtin_amdgcn_global_load_async_to_lds_b128)
#define ASYNC_CP 1
#else
#define ASYNC_CP 0
#endif

__device__ __forceinline__ float sigmoidf_(float x) { return 1.f / (1.f + __expf(-x)); }
__device__ __forceinline__ float softplusf_(float x) { return x > 20.f ? x : log1pf(__expf(x)); }

__device__ __forceinline__ v8f wmma_bf16(v16bf a, v16bf b, v8f c) {
  return __builtin_amdgcn_wmma_f32_16x16x32_bf16(false, a, false, b, (short)0, c, false, false);
}

// ---- 128-bit fragment loads (CDNA5 ISA 7.12.2 layouts) ----
// A fragment: per-lane two contiguous 8-elem runs of row m.  rowp = base + m*ld.
__device__ __forceinline__ v16bf load_a_frag_v(const bf16_t* rowp, int k0, int lane) {
  int khalf = (lane >> 4) << 3;
  v8bf lo = *(const v8bf*)(rowp + k0 + khalf);
  v8bf hi = *(const v8bf*)(rowp + k0 + khalf + 16);
  return __builtin_shufflevector(lo, hi, 0,1,2,3,4,5,6,7,8,9,10,11,12,13,14,15);
}
// B fragment: per-lane two contiguous 8-elem runs of row n (B stored [N][K]).
__device__ __forceinline__ v16bf load_b_frag_v(const bf16_t* rowp, int k0, int lane) {
  int kb = (lane >> 4) << 3;
  v8bf lo = *(const v8bf*)(rowp + k0 + kb);
  v8bf hi = *(const v8bf*)(rowp + k0 + kb + 16);
  return __builtin_shufflevector(lo, hi, 0,1,2,3,4,5,6,7,8,9,10,11,12,13,14,15);
}
// coalesced 16B row copy (global_load_b128 -> ds_store_b128); n multiple of 8
__device__ __forceinline__ void copy16(bf16_t* dst, const bf16_t* src, int n) {
  const uint4* s = (const uint4*)src;
  uint4* d = (uint4*)dst;
#pragma unroll
  for (int i = 0; i < n / 8; ++i) d[i] = s[i];
}
// async variant: one global_load_async_to_lds_b128 per 16 B, no VGPR round-trip
__device__ __forceinline__ void cp16_async(bf16_t* dst, const bf16_t* src, int n) {
#if ASYNC_CP
  AS3 v4i_* l = (AS3 v4i_*)dst;
  AS1 v4i_* g = (AS1 v4i_*)const_cast<bf16_t*>(src);
#pragma unroll
  for (int i = 0; i < n / 8; ++i)
    __builtin_amdgcn_global_load_async_to_lds_b128(g + i, l + i, 0, 0);
#else
  copy16(dst, src, n);
#endif
}
__device__ __forceinline__ void async_fence() {
#if ASYNC_CP
 #if __has_builtin(__builtin_amdgcn_s_wait_asynccnt)
  __builtin_amdgcn_s_wait_asynccnt(0);
 #else
  asm volatile("s_wait_asynccnt 0" ::: "memory");
 #endif
#endif
}

// ---------------- kernel 1: causal conv + SiLU + scatter (plus transposed copies) ----------------
__global__ void k_conv(const float* __restrict__ zx, const float* __restrict__ w,
                       const float* __restrict__ cb, const float* __restrict__ dtb,
                       float* __restrict__ xs, bf16_t* __restrict__ xdtT,
                       bf16_t* __restrict__ Bm, bf16_t* __restrict__ BmT,
                       bf16_t* __restrict__ Cm) {
  long idx = (long)blockIdx.x * blockDim.x + threadIdx.x;
  long total = (long)BD * LD * CONVDIM;
  if (idx >= total) return;
  int c = (int)(idx % CONVDIM);
  long bl = idx / CONVDIM;
  int l = (int)(bl % LD);
  int b = (int)(bl / LD);
  float acc = cb[c];
#pragma unroll
  for (int k = 0; k < DCONV; ++k) {
    int ll = l - (DCONV - 1) + k;
    if (ll >= 0) acc += w[c * DCONV + k] * zx[((long)b * LD + ll) * DIN + DINNER + c];
  }
  float s = acc * sigmoidf_(acc);          // SiLU
  if (c < DINNER) {
    int h = c >> 6, p = c & 63;
    float dtraw = zx[bl * DIN + (DIN - NHEADS) + h];
    float dt = fminf(softplusf_(dtraw + dtb[h]), 100.f);
    xs[bl * DINNER + c] = s;                                   // D-skip input
    xdtT[(((long)b * NHEADS + h) * HEADDIM + p) * LD + l] = (bf16_t)(s * dt); // [b][h][p][l]
  } else if (c < DINNER + DSTATE) {
    int n = c - DINNER;
    Bm[bl * DSTATE + n] = (bf16_t)s;                           // [l][n]
    BmT[((long)b * DSTATE + n) * LD + l] = (bf16_t)s;          // [b][n][l]
  } else {
    Cm[bl * DSTATE + (c - DINNER - DSTATE)] = (bf16_t)s;       // [l][n]
  }
}

// ---------------- kernel 2: a = dt*A, per-chunk inclusive scan ----------------
__global__ void k_dtscan(const float* __restrict__ zx, const float* __restrict__ dtb,
                         const float* __restrict__ Alog,
                         float* __restrict__ acs, float* __restrict__ csum) {
  __shared__ float sb[CHUNK];
  int t = threadIdx.x;
  int idx = blockIdx.x;                    // (b*NHEADS + h)*NCHUNK + c
  int c = idx % NCHUNK;
  int bh = idx / NCHUNK;
  int h = bh % NHEADS;
  int b = bh / NHEADS;
  int l = c * CHUNK + t;
  float dtraw = zx[((long)b * LD + l) * DIN + (DIN - NHEADS) + h];
  float dt = fminf(softplusf_(dtraw + dtb[h]), 100.f);
  float a = dt * (-__expf(Alog[h]));
  sb[t] = a;
  __syncthreads();
  for (int off = 1; off < CHUNK; off <<= 1) {
    float v = (t >= off) ? sb[t - off] : 0.f;
    __syncthreads();
    sb[t] += v;
    __syncthreads();
  }
  float cs = sb[t];
  acs[(long)bh * LD + l] = cs;
  if (t == CHUNK - 1) csum[bh * NCHUNK + c] = cs;
}

// ---------------- kernel 3: S(128x64) = B^T @ (decay .* x*dt) ----------------
__global__ void k_states(const bf16_t* __restrict__ BmT, const bf16_t* __restrict__ xdtT,
                         const float* __restrict__ acs, float* __restrict__ states) {
  extern __shared__ char smem[];
  bf16_t* At  = (bf16_t*)smem;                              // [128][KLD]  A op [n][l]
  bf16_t* Xt  = At + (size_t)DSTATE * KLD;                  // [64][KLD]   B op [p][l]
  float*  dec = (float*)(Xt + (size_t)HEADDIM * KLD);       // [256]
  int t = threadIdx.x, lane = t & 31, wave = t >> 5;
  int idx = blockIdx.x;                                     // ((b*NCHUNK)+c)*NHEADS + h
  int h = idx % NHEADS;
  int c = (idx / NHEADS) % NCHUNK;
  int b = idx / (NHEADS * NCHUNK);
  const float* acsr = acs + ((long)b * NHEADS + h) * LD + (long)c * CHUNK;
  float alast = acsr[CHUNK - 1];
  dec[t] = __expf(alast - acsr[t]);
  { // A: 128 rows x 256 bf16, async straight into LDS
    int n = t >> 1, half = t & 1;
    cp16_async(At + n * KLD + half * 128,
               BmT + ((long)b * DSTATE + n) * LD + (long)c * CHUNK + half * 128, 128);
  }
  async_fence();
  __syncthreads();                                          // dec ready, A landed
  { // X: 64 rows x 256 bf16, scale by per-column decay
    int p = t >> 2, q = t & 3;
    const bf16_t* src = xdtT + (((long)b * NHEADS + h) * HEADDIM + p) * LD + (long)c * CHUNK + q * 64;
    bf16_t* d = Xt + p * KLD + q * 64;
    const float* dv = dec + q * 64;
#pragma unroll
    for (int i = 0; i < 8; ++i) {
      v8bf v = ((const v8bf*)src)[i];
      v8bf o;
#pragma unroll
      for (int j = 0; j < 8; ++j) o[j] = (bf16_t)((float)v[j] * dv[i * 8 + j]);
      ((v8bf*)d)[i] = o;
    }
  }
  __syncthreads();
  int mt = wave;                                            // 8 m-tiles (state dim)
  const bf16_t* arow = At + (mt * 16 + (lane & 15)) * KLD;
  v16bf af[8];
#pragma unroll
  for (int kk = 0; kk < 8; ++kk) af[kk] = load_a_frag_v(arow, kk * 32, lane);
  float* outp = states + ((long)((b * NCHUNK + c) * NHEADS) + h) * (DSTATE * HEADDIM);
#pragma unroll
  for (int nt = 0; nt < 4; ++nt) {
    const bf16_t* brow = Xt + (nt * 16 + (lane & 15)) * KLD;
    v8f acc = {};
#pragma unroll
    for (int kk = 0; kk < 8; ++kk)
      acc = wmma_bf16(af[kk], load_b_frag_v(brow, kk * 32, lane), acc);
#pragma unroll
    for (int r = 0; r < 8; ++r) {
      int m = mt * 16 + r + ((lane >> 4) << 3);
      int n = nt * 16 + (lane & 15);
      outp[m * HEADDIM + n] = acc[r];                       // states [n][p]
    }
  }
}

// ---------------- kernel 4: recurrence P <- e^{sum a} P + S; emit prev^T [p][n] ----------------
__global__ void k_chain(const float* __restrict__ states, const float* __restrict__ csum,
                        bf16_t* __restrict__ prevT) {
  int t = threadIdx.x;
  int h = blockIdx.x % NHEADS;
  int b = blockIdx.x / NHEADS;
  float S[32];
#pragma unroll
  for (int j = 0; j < 32; ++j) S[j] = 0.f;
  for (int c = 0; c < NCHUNK; ++c) {
    long base = ((long)((b * NCHUNK + c) * NHEADS) + h) * (DSTATE * HEADDIM);
    float dec = __expf(csum[(b * NHEADS + h) * NCHUNK + c]);
#pragma unroll
    for (int j = 0; j < 32; ++j) {
      long e = t + 256 * j;                                 // e = n*64 + p
      int n = (int)(e >> 6), p = (int)(e & 63);
      prevT[base + (long)p * DSTATE + n] = (bf16_t)S[j];
      S[j] = dec * S[j] + states[base + e];
    }
  }
}

// ---------------- kernel 5: Y = rowscale(C@prev) + (C@B^T (.) L) @ x + x*D ----------------
__global__ void k_out(const bf16_t* __restrict__ Cm, const bf16_t* __restrict__ Bm,
                      const bf16_t* __restrict__ xdtT, const bf16_t* __restrict__ prevT,
                      const float* __restrict__ acs, const float* __restrict__ xs,
                      const float* __restrict__ Dp, float* __restrict__ out) {
  extern __shared__ char smem[];
  bf16_t* Cl = (bf16_t*)smem;                               // [256][NLD]  A op [l][n]
  bf16_t* Bl = Cl + (size_t)CHUNK * NLD;                    // [256][NLD]  B op for G: [j][n]
  bf16_t* Xt = Bl + (size_t)CHUNK * NLD;                    // [64][KLD]   B op: [p][j]
  bf16_t* Pt = Xt + (size_t)HEADDIM * KLD;                  // [64][NLD]   B op: [p][n]
  bf16_t* Gl = Pt + (size_t)HEADDIM * NLD;                  // [8 waves][16][GLD]
  float*  As = (float*)(Gl + (size_t)8 * 16 * GLD);         // [256]
  int t = threadIdx.x, lane = t & 31, wave = t >> 5;
  int idx = blockIdx.x;
  int h = idx % NHEADS;
  int c = (idx / NHEADS) % NCHUNK;
  int b = idx / (NHEADS * NCHUNK);
  long l0 = (long)b * LD + (long)c * CHUNK;
  { // cooperative stage: all pure copies go async straight into LDS
    int r = t;
    __builtin_prefetch(Cm + (l0 + r) * DSTATE, 0, 0);
    cp16_async(Cl + r * NLD, Cm + (l0 + r) * DSTATE, DSTATE);
    cp16_async(Bl + r * NLD, Bm + (l0 + r) * DSTATE, DSTATE);
    int p = t >> 2, q = t & 3;
    cp16_async(Xt + p * KLD + q * 64,
               xdtT + (((long)b * NHEADS + h) * HEADDIM + p) * LD + (long)c * CHUNK + q * 64, 64);
    if (t < 128) {
      int pp = t >> 1, half = t & 1;
      cp16_async(Pt + pp * NLD + half * 64,
                 prevT + ((long)((b * NCHUNK + c) * NHEADS) + h) * (DSTATE * HEADDIM)
                       + (long)pp * DSTATE + half * 64, 64);
    }
    As[r] = acs[((long)b * NHEADS + h) * LD + (long)c * CHUNK + r];
  }
  async_fence();
  __syncthreads();
  float Dh = Dp[h];
  bf16_t* Gw = Gl + (size_t)wave * 16 * GLD;
  for (int mi = 0; mi < 2; ++mi) {
    int mt = wave * 2 + mi;                                 // 16 row-tiles over 8 waves
    const bf16_t* crow = Cl + (mt * 16 + (lane & 15)) * NLD;
    v16bf ca[4];
#pragma unroll
    for (int kk = 0; kk < 4; ++kk) ca[kk] = load_a_frag_v(crow, kk * 32, lane); // reuse C frags
    v8f acc[4] = {};
    // ---- Y_off = C @ prev, then rowscale exp(a_cs) ----
#pragma unroll
    for (int kk = 0; kk < 4; ++kk)
#pragma unroll
      for (int nt = 0; nt < 4; ++nt)
        acc[nt] = wmma_bf16(ca[kk],
                            load_b_frag_v(Pt + (nt * 16 + (lane & 15)) * NLD, kk * 32, lane),
                            acc[nt]);
#pragma unroll
    for (int nt = 0; nt < 4; ++nt)
#pragma unroll
      for (int r = 0; r < 8; ++r) {
        int i = mt * 16 + r + ((lane >> 4) << 3);
        acc[nt][r] *= __expf(As[i]);
      }
    // ---- Y_diag: causal 32-column blocks ----
    for (int jp = 0; jp <= (mt >> 1); ++jp) {
#pragma unroll
      for (int sub = 0; sub < 2; ++sub) {
        int jt = jp * 2 + sub;
        const bf16_t* brow = Bl + (jt * 16 + (lane & 15)) * NLD;
        v8f g = {};
#pragma unroll
        for (int kk = 0; kk < 4; ++kk)
          g = wmma_bf16(ca[kk], load_b_frag_v(brow, kk * 32, lane), g);
#pragma unroll
        for (int r = 0; r < 8; ++r) {                       // mask + decay -> bf16 A stage
          int i = mt * 16 + r + ((lane >> 4) << 3);
          int j = jt * 16 + (lane & 15);
          float v = (j <= i) ? g[r] * __expf(As[i] - As[j]) : 0.f;
          Gw[(r + ((lane >> 4) << 3)) * GLD + sub * 16 + (lane & 15)] = (bf16_t)v;
        }
      }
      v16bf ag = load_a_frag_v(Gw + (lane & 15) * GLD, 0, lane); // per-wave LDS, in-order DS
#pragma unroll
      for (int nt = 0; nt < 4; ++nt)
        acc[nt] = wmma_bf16(ag,
                            load_b_frag_v(Xt + (nt * 16 + (lane & 15)) * KLD, jp * 32, lane),
                            acc[nt]);
    }
    // ---- D-skip + store ----
#pragma unroll
    for (int nt = 0; nt < 4; ++nt)
#pragma unroll
      for (int r = 0; r < 8; ++r) {
        int i = mt * 16 + r + ((lane >> 4) << 3);
        int p = nt * 16 + (lane & 15);
        long gl = l0 + i;
        float xv = xs[gl * DINNER + h * HEADDIM + p];
        out[gl * DINNER + h * HEADDIM + p] = acc[nt][r] + xv * Dh;
      }
  }
}

// ---------------- kernel 6: y *= silu(z); RMSNorm (in place on d_out) ----------------
__global__ void k_norm(const float* __restrict__ zx, const float* __restrict__ nw,
                       float* __restrict__ out) {
  __shared__ float red[256];
  long row = blockIdx.x;                   // b*L + l
  int t = threadIdx.x;
  float yv[8];
  float ss = 0.f;
#pragma unroll
  for (int k = 0; k < 8; ++k) {
    int d = t + 256 * k;
    float y = out[row * DINNER + d];
    float z = zx[row * DIN + d];
    float v = y * (z * sigmoidf_(z));
    yv[k] = v;
    ss += v * v;
  }
  red[t] = ss;
  __syncthreads();
  for (int off = 128; off > 0; off >>= 1) {
    if (t < off) red[t] += red[t + off];
    __syncthreads();
  }
  float rstd = rsqrtf(red[0] / (float)DINNER + EPSV);
#pragma unroll
  for (int k = 0; k < 8; ++k) {
    int d = t + 256 * k;
    out[row * DINNER + d] = yv[k] * rstd * nw[d];
  }
}

// ---------------- host launch ----------------
extern "C" void kernel_launch(void* const* d_in, const int* in_sizes, int n_in,
                              void* d_out, int out_size, void* d_ws, size_t ws_size,
                              hipStream_t stream) {
  (void)in_sizes; (void)n_in; (void)out_size; (void)ws_size;
  const float* zx  = (const float*)d_in[0];
  const float* w   = (const float*)d_in[1];
  const float* cb  = (const float*)d_in[2];
  const float* dtb = (const float*)d_in[3];
  const float* Al  = (const float*)d_in[4];
  const float* Dp  = (const float*)d_in[5];
  const float* nw  = (const float*)d_in[6];
  float* out = (float*)d_out;

  char* ws = (char*)d_ws;
  size_t off = 0;
  auto alloc = [&](size_t bytes) { size_t o = off; off = (off + bytes + 255) & ~(size_t)255; return o; };
  float*  xs     = (float*) (ws + alloc((size_t)BD * LD * DINNER * 4));
  bf16_t* xdtT   = (bf16_t*)(ws + alloc((size_t)BD * LD * DINNER * 2));
  bf16_t* Bm     = (bf16_t*)(ws + alloc((size_t)BD * LD * DSTATE * 2));
  bf16_t* BmT    = (bf16_t*)(ws + alloc((size_t)BD * LD * DSTATE * 2));
  bf16_t* Cm     = (bf16_t*)(ws + alloc((size_t)BD * LD * DSTATE * 2));
  float*  acs    = (float*) (ws + alloc((size_t)BD * NHEADS * LD * 4));
  float*  csum   = (float*) (ws + alloc((size_t)BD * NHEADS * NCHUNK * 4));
  float*  states = (float*) (ws + alloc((size_t)BD * NCHUNK * NHEADS * DSTATE * HEADDIM * 4));
  bf16_t* prevT  = (bf16_t*)(ws + alloc((size_t)BD * NCHUNK * NHEADS * DSTATE * HEADDIM * 2));

  long n1 = (long)BD * LD * CONVDIM;
  k_conv<<<(unsigned)((n1 + 255) / 256), 256, 0, stream>>>(zx, w, cb, dtb, xs, xdtT, Bm, BmT, Cm);
  k_dtscan<<<BD * NHEADS * NCHUNK, CHUNK, 0, stream>>>(zx, dtb, Al, acs, csum);

  size_t lds3 = (size_t)DSTATE * KLD * 2 + (size_t)HEADDIM * KLD * 2 + (size_t)CHUNK * 4; // ~100 KB
  k_states<<<BD * NCHUNK * NHEADS, 256, lds3, stream>>>(BmT, xdtT, acs, states);

  k_chain<<<BD * NHEADS, 256, 0, stream>>>(states, csum, prevT);

  size_t lds5 = (size_t)CHUNK * NLD * 2 * 2 + (size_t)HEADDIM * KLD * 2
              + (size_t)HEADDIM * NLD * 2 + (size_t)8 * 16 * GLD * 2
              + (size_t)CHUNK * 4;                                                        // ~197 KB
  k_out<<<BD * NCHUNK * NHEADS, 256, lds5, stream>>>(Cm, Bm, xdtT, prevT, acs, xs, Dp, out);

  k_norm<<<BD * LD, 256, 0, stream>>>(zx, nw, out);
}
